// WindowBasedSelfAttention_11089605558962
// MI455X (gfx1250) — matrix operand order repaired and checked by reading, not compiled, not run on hardware
//
#include <hip/hip_runtime.h>
#include <hip/hip_bf16.h>
#include <math.h>

typedef __attribute__((ext_vector_type(16))) __bf16 v16bf;
typedef __attribute__((ext_vector_type(8)))  float  v8f;

#define BATCH 2
#define CH    32
#define WSZ   64
#define NTOK  64          // 8x8 windows per batch
#define DIN   4096
#define DMID  1024
#define PITCH 65          // LDS row pitch (float2) to dodge bank conflicts

__device__ __forceinline__ int brev6(int x) { return (int)(__brev((unsigned)x) >> 26); }

__device__ __forceinline__ float2 cmul(float2 a, float2 b) {
  return make_float2(a.x * b.x - a.y * b.y, a.x * b.y + a.y * b.x);
}

// ---- gfx1250 async global->LDS copy of one 64x64 f32 window (16KB) --------
// Rows are 256B apart in LDS-staging (contiguous), 2048B apart in global.
// Uses GLOBAL_LOAD_ASYNC_TO_LDS_B128 in GVS mode (SGPR64 base + VGPR32 offset).
// Flat shared-pointer bits [31:0] are the LDS byte offset (CDNA5 aperture map).
__device__ __forceinline__ void async_load_window(const float* xs, float* stage, int tid)
{
  unsigned long long base = (unsigned long long)(uintptr_t)xs;
  unsigned ldsBase = (unsigned)(uintptr_t)stage;
  #pragma unroll
  for (int it = 0; it < 4; ++it) {
    int p = it * 256 + tid;                         // 1024 x b128 transfers
    unsigned gOff = (unsigned)(((p >> 4) << 11) + ((p & 15) << 4));
    unsigned lOff = ldsBase + ((unsigned)p << 4);
    asm volatile("global_load_async_to_lds_b128 %0, %1, %2"
                 :: "v"(lOff), "v"(gOff), "s"(base) : "memory");
  }
}
__device__ __forceinline__ void wait_async0() {
  asm volatile("s_wait_asynccnt 0x0" ::: "memory");
}

// In-place 64x64 2D FFT over LDS (pitch PITCH). sign=-1 fwd, +1 inverse.
// Applies "ortho" scale 1/64. 256 threads cooperate. Ends after a barrier.
__device__ void fft2_64(float2* buf, int tid, float sign)
{
  // bit-reverse permute columns (in place; pair handled by the lower index)
  for (int p = tid; p < WSZ * WSZ; p += 256) {
    int r = p >> 6, c = p & 63, cb = brev6(c);
    if (c < cb) {
      float2 t0 = buf[r * PITCH + c];
      buf[r * PITCH + c]  = buf[r * PITCH + cb];
      buf[r * PITCH + cb] = t0;
    }
  }
  __syncthreads();
  // DIT stages along columns
  for (int s = 0; s < 6; ++s) {
    int h = 1 << s;
    for (int p = tid; p < 2048; p += 256) {
      int r = p >> 5, t = p & 31;
      int j  = t & (h - 1);
      int i0 = ((t >> s) << (s + 1)) + j;
      int i1 = i0 + h;
      float ang = sign * 3.14159265358979f * (float)j / (float)h;
      float sn, cs; __sincosf(ang, &sn, &cs);
      float2 u = buf[r * PITCH + i0];
      float2 v = buf[r * PITCH + i1];
      float2 tw = make_float2(v.x * cs - v.y * sn, v.x * sn + v.y * cs);
      buf[r * PITCH + i0] = make_float2(u.x + tw.x, u.y + tw.y);
      buf[r * PITCH + i1] = make_float2(u.x - tw.x, u.y - tw.y);
    }
    __syncthreads();
  }
  // bit-reverse permute rows (in place)
  for (int p = tid; p < WSZ * WSZ; p += 256) {
    int r = p >> 6, c = p & 63, rb = brev6(r);
    if (r < rb) {
      float2 t0 = buf[r * PITCH + c];
      buf[r * PITCH + c]  = buf[rb * PITCH + c];
      buf[rb * PITCH + c] = t0;
    }
  }
  __syncthreads();
  // DIT stages along rows
  for (int s = 0; s < 6; ++s) {
    int h = 1 << s;
    for (int p = tid; p < 2048; p += 256) {
      int c = p >> 5, t = p & 31;
      int j  = t & (h - 1);
      int i0 = ((t >> s) << (s + 1)) + j;
      int i1 = i0 + h;
      float ang = sign * 3.14159265358979f * (float)j / (float)h;
      float sn, cs; __sincosf(ang, &sn, &cs);
      float2 u = buf[i0 * PITCH + c];
      float2 v = buf[i1 * PITCH + c];
      float2 tw = make_float2(v.x * cs - v.y * sn, v.x * sn + v.y * cs);
      buf[i0 * PITCH + c] = make_float2(u.x + tw.x, u.y + tw.y);
      buf[i1 * PITCH + c] = make_float2(u.x - tw.x, u.y - tw.y);
    }
    __syncthreads();
  }
  // ortho scaling 1/sqrt(64*64)
  for (int p = tid; p < WSZ * WSZ; p += 256) {
    int r = p >> 6, c = p & 63;
    float2 z = buf[r * PITCH + c];
    buf[r * PITCH + c] = make_float2(z.x * 0.015625f, z.y * 0.015625f);
  }
  __syncthreads();
}

// ---------------- Kernel 1: FFT all 32 channels of one window, channel argmax,
// capture the 32 "firstrow" spectrum values that seed the tokens.
// Channel loop is double-buffered: async DMA of channel c+1 overlaps FFT of c.
__global__ __launch_bounds__(256) void k_fft_argmax(
    const float* __restrict__ x, unsigned char* __restrict__ idx8,
    float2* __restrict__ firstrow)
{
  __shared__ float2 sA[WSZ * PITCH];
  __shared__ float  stage[2][WSZ * WSZ];
  const int tid = threadIdx.x;
  const int nh = blockIdx.x >> 3, nw = blockIdx.x & 7;
  const int b  = blockIdx.y;

  float bestm[16]; int bestc[16];
  #pragma unroll
  for (int u = 0; u < 16; ++u) { bestm[u] = -1.0f; bestc[u] = 0; }

  const float* xwin = x + ((size_t)b * CH * 512 + (size_t)nh * 64) * 512 + nw * 64;
  // prologue: start DMA of channel 0
  async_load_window(xwin, stage[0], tid);

  for (int ch = 0; ch < CH; ++ch) {
    wait_async0();
    __syncthreads();                       // stage[ch&1] now holds channel ch
    if (ch + 1 < CH)                       // overlap next DMA with this FFT
      async_load_window(xwin + (size_t)(ch + 1) * 512 * 512, stage[(ch + 1) & 1], tid);
    const float* st = stage[ch & 1];
    for (int p = tid; p < WSZ * WSZ; p += 256) {
      int i = p >> 6, j = p & 63;
      sA[i * PITCH + j] = make_float2(st[p], 0.0f);
    }
    __syncthreads();
    fft2_64(sA, tid, -1.0f);
    #pragma unroll
    for (int u = 0; u < 16; ++u) {
      int p = u * 256 + tid;
      int i = p >> 6, j = p & 63;
      float2 z = sA[i * PITCH + j];
      float m2 = z.x * z.x + z.y * z.y;
      if (m2 > bestm[u]) { bestm[u] = m2; bestc[u] = ch; }
      // tokens source: shifted ff[b,0] flat[0..32) = F[b^1, c=16, win(4,4), row 32, col 32..63]
      if (ch == 16 && nh == 4 && nw == 4 && i == 32 && j >= 32)
        firstrow[(b ^ 1) * 32 + (j - 32)] = z;
    }
    __syncthreads();
  }
  // write argmax channel (in the fftshift-ed coordinate frame) as uint8
  const int tSh = ((nh + 4) & 7) * 8 + ((nw + 4) & 7);
  const int bSh = b ^ 1;
  #pragma unroll
  for (int u = 0; u < 16; ++u) {
    int p = u * 256 + tid;
    int i = p >> 6, j = p & 63;
    int fSh = (((i + 32) & 63) << 6) | ((j + 32) & 63);
    idx8[((size_t)(bSh * NTOK + tSh)) * DIN + fSh] = (unsigned char)((bestc[u] + 16) & 31);
  }
}

// ---------------- Kernel 2: tokens[b,t,f] = firstrow[b][idx] ----------------
__global__ __launch_bounds__(256) void k_tokens(
    const unsigned char* __restrict__ idx8, const float2* __restrict__ firstrow,
    float2* __restrict__ tokens)
{
  int g = blockIdx.x * 256 + threadIdx.x;   // 0 .. 2*64*4096
  int b = g >> 18;
  tokens[g] = firstrow[b * 32 + idx8[g]];
}

// ---------------- Kernel 3: generic complex GEMM on bf16 WMMA ---------------
// C[b,m,n] = sum_k A[b,m,k] * B[k,n] (+bias[n]) ; M fixed = 64 tokens.
// bWeight=1: B[k,n] = W[n*K+k] (row-contiguous weight rows)
// bWeight=0: B[k,n] = Bsrc[k*N+n] (natural activation layout)
// aReal=1:   A is real f32 (softmax weights), imag treated as 0.
__global__ __launch_bounds__(32) void k_cgemm_wmma(
    const float2* __restrict__ Ac, const float* __restrict__ Ar32,
    const float2* __restrict__ Bsrc, float2* __restrict__ Cc,
    const float2* __restrict__ bias,
    int K, int N, int act, int aReal, int bWeight, long long bBatchStride)
{
  const int lane = threadIdx.x;
  const int half = lane >> 4, lr = lane & 15;
  const int nBase = blockIdx.x * 16;
  const int mBase = blockIdx.y * 16;
  const int b = blockIdx.z;
  const size_t aOff = (size_t)b * 64 * K;
  const size_t cOff = (size_t)b * 64 * N;
  const size_t bOff = (size_t)b * (size_t)bBatchStride;

  v8f accRe = {0.f, 0.f, 0.f, 0.f, 0.f, 0.f, 0.f, 0.f};
  v8f accIm = {0.f, 0.f, 0.f, 0.f, 0.f, 0.f, 0.f, 0.f};

  for (int kb = 0; kb < K; kb += 32) {
    v16bf ar, ai, ain, br, bi;
    // ---- A fragment: lane<16 holds M=lr, K=0..15; lane>=16: K=16..31
    if (!aReal) {
      const float2* arow = Ac + aOff + (size_t)(mBase + lr) * K + kb + (half << 4);
      #pragma unroll
      for (int e = 0; e < 8; ++e) {
        float2 v0 = arow[e];
        float2 v1 = arow[8 + e];
        ar[e]     = (__bf16)v0.x;  ai[e]     = (__bf16)v0.y;  ain[e]     = (__bf16)(-v0.y);
        ar[8 + e] = (__bf16)v1.x;  ai[8 + e] = (__bf16)v1.y;  ain[8 + e] = (__bf16)(-v1.y);
      }
    } else {
      const float* arow = Ar32 + aOff + (size_t)(mBase + lr) * K + kb + (half << 4);
      #pragma unroll
      for (int e = 0; e < 8; ++e) {
        ar[e]     = (__bf16)arow[e];
        ar[8 + e] = (__bf16)arow[8 + e];
        ai[e] = (__bf16)0.f; ai[8 + e] = (__bf16)0.f;
        ain[e] = (__bf16)0.f; ain[8 + e] = (__bf16)0.f;
      }
    }
    // ---- B fragment: lane<16 holds N=lr, K=0..15 of the 32-block; lane>=16: K=16..31
    if (bWeight) {
      const float2* wrow = Bsrc + bOff + (size_t)(nBase + lr) * K + kb + (half << 4);
      __builtin_prefetch(wrow + 32, 0, 1);   // global_prefetch next K-tile
      #pragma unroll
      for (int e = 0; e < 16; ++e) {
        float2 v = wrow[e];
        br[e] = (__bf16)v.x;  bi[e] = (__bf16)v.y;
      }
    } else {
      const float2* bcol = Bsrc + bOff + (size_t)(kb + (half << 4)) * N + (nBase + lr);
      #pragma unroll
      for (int e = 0; e < 16; ++e) {
        float2 v = bcol[(size_t)e * N];
        br[e] = (__bf16)v.x;  bi[e] = (__bf16)v.y;
      }
    }
    // complex MAC: Re += Ar*Br + (-Ai)*Bi ; Im += Ar*Bi + Ai*Br
    accRe = __builtin_amdgcn_wmma_f32_16x16x32_bf16(false, ar,  false, br, (short)0, accRe, false, false);
    accIm = __builtin_amdgcn_wmma_f32_16x16x32_bf16(false, ar,  false, bi, (short)0, accIm, false, false);
    if (!aReal) {
      accRe = __builtin_amdgcn_wmma_f32_16x16x32_bf16(false, ain, false, bi, (short)0, accRe, false, false);
      accIm = __builtin_amdgcn_wmma_f32_16x16x32_bf16(false, ai,  false, br, (short)0, accIm, false, false);
    }
  }

  // ---- epilogue: C/D layout: lane -> col (lr), VGPR r -> row (half*8 + r)
  const int n = nBase + lr;
  float2 bv = bias ? bias[n] : make_float2(0.f, 0.f);
  #pragma unroll
  for (int r = 0; r < 8; ++r) {
    int m = mBase + (half << 3) + r;
    float re = accRe[r] + bv.x;
    float im = accIm[r] + bv.y;
    if (act == 1) {  // complex leaky-relu, slope 0.01 on re/im independently
      re = re > 0.f ? re : 0.01f * re;
      im = im > 0.f ? im : 0.01f * im;
    }
    Cc[cOff + (size_t)m * N + n] = make_float2(re, im);
  }
}

// ---------------- Kernel 4: row softmax of |scores|/32 ----------------------
__global__ __launch_bounds__(64) void k_softmax_abs(
    const float2* __restrict__ scores, float* __restrict__ aw)
{
  __shared__ float red[64];
  const int s = blockIdx.x, b = blockIdx.y, t = threadIdx.x;
  float2 z = scores[((size_t)(b * NTOK + s)) * NTOK + t];
  float m = sqrtf(z.x * z.x + z.y * z.y) * (1.0f / 32.0f);  // / sqrt(D_MID)
  red[t] = m; __syncthreads();
  for (int o = 32; o > 0; o >>= 1) { if (t < o) red[t] = fmaxf(red[t], red[t + o]); __syncthreads(); }
  float mx = red[0]; __syncthreads();
  float e = __expf(m - mx);
  red[t] = e; __syncthreads();
  for (int o = 32; o > 0; o >>= 1) { if (t < o) red[t] = red[t] + red[t + o]; __syncthreads(); }
  aw[((size_t)(b * NTOK + s)) * NTOK + t] = e / red[0];
}

// ---------------- Kernel 5: fused FFT -> spectral filter -> iFFT -> real ----
__global__ __launch_bounds__(256) void k_apply_ifft(
    const float* __restrict__ x, const float2* __restrict__ attn,
    float* __restrict__ out)
{
  __shared__ float2 sA[WSZ * PITCH];
  __shared__ float  stage[WSZ * WSZ];
  const int tid = threadIdx.x;
  const int nh = blockIdx.x >> 3, nw = blockIdx.x & 7;
  const int c = blockIdx.y, b = blockIdx.z;

  const float* xs = x + (((size_t)(b * CH + c) * 512) + (size_t)nh * 64) * 512 + nw * 64;
  async_load_window(xs, stage, tid);       // DMA window into LDS
  wait_async0();
  __syncthreads();
  for (int p = tid; p < WSZ * WSZ; p += 256) {
    int i = p >> 6, j = p & 63;
    sA[i * PITCH + j] = make_float2(stage[p], 0.0f);
  }
  __syncthreads();
  fft2_64(sA, tid, -1.0f);
  // fftshift/ifftshift cancel on F; only the filter index is shifted (all axes)
  const int tSh = ((nh + 4) & 7) * 8 + ((nw + 4) & 7);
  const float2* arow = attn + ((size_t)((b ^ 1) * NTOK + tSh)) * DIN;
  for (int p = tid; p < WSZ * WSZ; p += 256) {
    int i = p >> 6, j = p & 63;
    int fSh = (((i + 32) & 63) << 6) | ((j + 32) & 63);
    sA[i * PITCH + j] = cmul(sA[i * PITCH + j], arow[fSh]);
  }
  __syncthreads();
  fft2_64(sA, tid, +1.0f);
  float* os = out + (((size_t)(b * CH + c) * 512) + (size_t)nh * 64) * 512 + nw * 64;
  for (int p = tid; p < WSZ * WSZ; p += 256) {
    int i = p >> 6, j = p & 63;
    os[(size_t)i * 512 + j] = sA[i * PITCH + j].x;
  }
}

// ---------------- Kernel 6: filter output (complex, un-shifted) -------------
__global__ __launch_bounds__(256) void k_filt(
    const float2* __restrict__ attn, float* __restrict__ outf)
{
  int g = blockIdx.x * 256 + threadIdx.x;    // 0 .. 2*512*512
  int b = g >> 18, rem = g & 262143;
  int y = rem >> 9, xx = rem & 511;
  int nh = y >> 6, i = y & 63, nw = xx >> 6, j = xx & 63;
  float2 v = attn[((size_t)(b * NTOK + nh * 8 + nw)) * DIN + ((i << 6) | j)];
  outf[2 * g]     = v.x;
  outf[2 * g + 1] = v.y;
}

extern "C" void kernel_launch(void* const* d_in, const int* in_sizes, int n_in,
                              void* d_out, int out_size, void* d_ws, size_t ws_size,
                              hipStream_t stream)
{
  (void)in_sizes; (void)n_in; (void)out_size;
  const float*  x  = (const float*) d_in[0];
  const float2* w1 = (const float2*)d_in[1];
  const float2* b1 = (const float2*)d_in[2];
  const float2* wq = (const float2*)d_in[3];
  const float2* bq = (const float2*)d_in[4];
  const float2* wk = (const float2*)d_in[5];
  const float2* bk = (const float2*)d_in[6];
  const float2* wv = (const float2*)d_in[7];
  const float2* bvb= (const float2*)d_in[8];
  const float2* w2 = (const float2*)d_in[9];
  const float2* b2 = (const float2*)d_in[10];
  float* out = (float*)d_out;

  char* w = (char*)d_ws;
  size_t cur = 0;
  auto alloc = [&](size_t bytes) -> void* {
    void* p = w + cur;
    cur += bytes;
    cur = (cur + 255) & ~(size_t)255;
    return p;
  };
  unsigned char* idx8 = (unsigned char*)alloc((size_t)BATCH * NTOK * DIN);
  float2* firstrow = (float2*)alloc(64 * sizeof(float2));
  float2* tokens = (float2*)alloc((size_t)BATCH * NTOK * DIN  * sizeof(float2));
  float2* yb     = (float2*)alloc((size_t)BATCH * NTOK * DMID * sizeof(float2));
  float2* qb     = (float2*)alloc((size_t)BATCH * NTOK * DMID * sizeof(float2));
  float2* kb     = (float2*)alloc((size_t)BATCH * NTOK * DMID * sizeof(float2));
  float2* vb     = (float2*)alloc((size_t)BATCH * NTOK * DMID * sizeof(float2));
  float2* sc     = (float2*)alloc((size_t)BATCH * NTOK * NTOK * sizeof(float2));
  float*  aw     = (float*) alloc((size_t)BATCH * NTOK * NTOK * sizeof(float));
  float2* y2     = (float2*)alloc((size_t)BATCH * NTOK * DMID * sizeof(float2));
  float2* attn   = (float2*)alloc((size_t)BATCH * NTOK * DIN  * sizeof(float2));
  if (cur > ws_size) return;  // not enough scratch (deterministic bail)

  // 1) per-window spectra + channel argmax + token seeds (async double-buffer)
  k_fft_argmax<<<dim3(64, BATCH), 256, 0, stream>>>(x, idx8, firstrow);
  // 2) gather tokens [B,64,4096]
  k_tokens<<<dim3((BATCH * NTOK * DIN) / 256), 256, 0, stream>>>(idx8, firstrow, tokens);
  // 3) y = leaky(tokens @ w1^T + b1)
  k_cgemm_wmma<<<dim3(DMID / 16, 4, BATCH), 32, 0, stream>>>(
      tokens, nullptr, w1, yb, b1, DIN, DMID, 1, 0, 1, 0);
  // 4) q, k, v
  k_cgemm_wmma<<<dim3(DMID / 16, 4, BATCH), 32, 0, stream>>>(
      yb, nullptr, wq, qb, bq, DMID, DMID, 0, 0, 1, 0);
  k_cgemm_wmma<<<dim3(DMID / 16, 4, BATCH), 32, 0, stream>>>(
      yb, nullptr, wk, kb, bk, DMID, DMID, 0, 0, 1, 0);
  k_cgemm_wmma<<<dim3(DMID / 16, 4, BATCH), 32, 0, stream>>>(
      yb, nullptr, wv, vb, bvb, DMID, DMID, 0, 0, 1, 0);
  // 5) scores = q @ k^T  (k rows play the "weight" role, per-batch stride)
  k_cgemm_wmma<<<dim3(NTOK / 16, 4, BATCH), 32, 0, stream>>>(
      qb, nullptr, kb, sc, nullptr, DMID, NTOK, 0, 0, 1, (long long)NTOK * DMID);
  // 6) softmax over |scores|/sqrt(D_MID)
  k_softmax_abs<<<dim3(NTOK, BATCH), 64, 0, stream>>>(sc, aw);
  // 7) y2 = attn_w(real) @ v
  k_cgemm_wmma<<<dim3(DMID / 16, 4, BATCH), 32, 0, stream>>>(
      nullptr, aw, vb, y2, nullptr, NTOK, DMID, 0, 1, 0, (long long)NTOK * DMID);
  // 8) attn = y2 @ w2^T + b2
  k_cgemm_wmma<<<dim3(DIN / 16, 4, BATCH), 32, 0, stream>>>(
      y2, nullptr, w2, attn, b2, DMID, DIN, 0, 0, 1, 0);
  // 9) fused FFT -> filter -> iFFT -> real image
  k_apply_ifft<<<dim3(64, CH, BATCH), 256, 0, stream>>>(x, attn, out);
  // 10) filter output (complex64 interleaved) after the image
  k_filt<<<dim3((BATCH * 512 * 512) / 256), 256, 0, stream>>>(
      attn, out + (size_t)BATCH * CH * 512 * 512);
}
